// MixtureOfExperts_12335146074163
// MI455X (gfx1250) — matrix operand order repaired and checked by reading, not compiled, not run on hardware
//
#include <hip/hip_runtime.h>
#include <hip/hip_bf16.h>

// ---------------- problem constants ----------------
#define NB 4
#define NS 2048
#define ND 768
#define NF 3072
#define NE 8
#define NTOK (NB * NS)          // 8192 tokens
#define LB_W 0.01f
#define Z_W  0.001f

// ---------------- types ----------------
typedef __attribute__((ext_vector_type(16))) __bf16        v16bf;
typedef __attribute__((ext_vector_type(8)))  float         v8f;
typedef __attribute__((ext_vector_type(4)))  float         v4f;
typedef __attribute__((ext_vector_type(4)))  int           v4i;
typedef __attribute__((ext_vector_type(4)))  unsigned int  v4u;
typedef __attribute__((ext_vector_type(2)))  unsigned int  v2u;

union FragBF { v16bf v; v4i q[2]; };

__device__ __forceinline__ unsigned short f2bf(float f) {
    unsigned int u = __float_as_uint(f);
    unsigned int r = u + 0x7fffu + ((u >> 16) & 1u);   // round-to-nearest-even
    return (unsigned short)(r >> 16);
}
__device__ __forceinline__ unsigned int f2bf2(float lo, float hi) {
    return (unsigned int)f2bf(lo) | ((unsigned int)f2bf(hi) << 16);
}

// async global->LDS b128 copy (CDNA5 TDM-adjacent path, tracked by ASYNCcnt).
// vdst VGPR = per-lane LDS byte address (generic-pointer low 32 bits), vaddr = 64b global addr.
__device__ __forceinline__ void async_copy_b128(const void* gptr, void* lptr) {
    unsigned lds = (unsigned)(size_t)lptr;
    unsigned long long g = (unsigned long long)(size_t)gptr;
    asm volatile("global_load_async_to_lds_b128 %0, %1, off" :: "v"(lds), "v"(g) : "memory");
}
__device__ __forceinline__ void wait_async() {
    asm volatile("s_wait_asynccnt 0x0" ::: "memory");
}

// LDS leading dims in ushorts: 40 (=80B) keeps 16B chunks aligned, spreads banks.
#define AS_LD 40
#define BS_LD 40

// A fragment 16x32 bf16: two contiguous 16B chunks per lane (ISA 7.12.2 layout).
__device__ __forceinline__ v16bf load_a_frag(const unsigned short* As, int row0, int lane) {
    const int half = lane >> 4, r = lane & 15;
    const unsigned short* p = As + (row0 + r) * AS_LD + half * 8;
    FragBF f;
    f.q[0] = *(const v4i*)(p);
    f.q[1] = *(const v4i*)(p + 16);
    return f.v;
}
// B fragment 32x16 bf16 from TRANSPOSED tile Bt[col][k]: two 16B chunks per lane (same k order as A).
__device__ __forceinline__ v16bf load_b_frag(const unsigned short* Bt, int col0, int lane) {
    const int half = lane >> 4, r = lane & 15;
    const unsigned short* p = Bt + (col0 + r) * BS_LD + half * 8;
    FragBF f;
    f.q[0] = *(const v4i*)(p);
    f.q[1] = *(const v4i*)(p + 16);
    return f.v;
}

// ---------------- small kernels ----------------

__global__ void moe_zero(float* __restrict__ p, int n) {
    int i = blockIdx.x * blockDim.x + threadIdx.x;
    if (i < n) p[i] = 0.0f;
}

// x fp32 -> bf16 (so GEMM A-fills become pure async bf16 copies)
__global__ void moe_cvt_x(const float* __restrict__ x, unsigned short* __restrict__ xbf, int nquads) {
    int i = blockIdx.x * blockDim.x + threadIdx.x;
    if (i < nquads) {
        v4f v = *(const v4f*)(x + (size_t)i * 4);
        *(v2u*)(xbf + (size_t)i * 4) = (v2u){f2bf2(v[0], v[1]), f2bf2(v[2], v[3])};
    }
}

// one wave32 per token: router logits, softmax, top-2, combine weights, loss terms
__global__ __launch_bounds__(256) void moe_router(const float* __restrict__ x,
                                                  const float* __restrict__ Wr,
                                                  float* __restrict__ probs,
                                                  float* __restrict__ combine,
                                                  float* __restrict__ lse2) {
    const int wid = threadIdx.x >> 5;
    const int lane = threadIdx.x & 31;
    const int t = blockIdx.x * 8 + wid;
    if (t >= NTOK) return;

    const float* xt = x + (size_t)t * ND;
    float acc[NE] = {};
    for (int d = lane; d < ND; d += 32) {
        float xv = xt[d];
        const float* wr = Wr + d * NE;
#pragma unroll
        for (int e = 0; e < NE; ++e) acc[e] += xv * wr[e];
    }
#pragma unroll
    for (int off = 16; off > 0; off >>= 1) {
#pragma unroll
        for (int e = 0; e < NE; ++e) acc[e] += __shfl_xor(acc[e], off, 32);
    }

    float m = acc[0];
#pragma unroll
    for (int e = 1; e < NE; ++e) m = fmaxf(m, acc[e]);
    float p[NE]; float se = 0.0f;
#pragma unroll
    for (int e = 0; e < NE; ++e) { p[e] = expf(acc[e] - m); se += p[e]; }
    float inv = 1.0f / se;
#pragma unroll
    for (int e = 0; e < NE; ++e) p[e] *= inv;
    float lse = m + logf(se);

    int i0 = 0; float p0 = p[0];
#pragma unroll
    for (int e = 1; e < NE; ++e) if (p[e] > p0) { p0 = p[e]; i0 = e; }
    int i1 = -1; float p1 = -1.0f;
#pragma unroll
    for (int e = 0; e < NE; ++e) if (e != i0 && p[e] > p1) { p1 = p[e]; i1 = e; }
    float denom = p0 + p1 + 1e-8f;
    float w0 = p0 / denom, w1 = p1 / denom;

    if (lane == 0) {
#pragma unroll
        for (int e = 0; e < NE; ++e) {
            probs[t * NE + e] = p[e];
            float c = 0.0f;
            if (e == i0) c = w0;
            if (e == i1) c = w1;
            combine[t * NE + e] = c;
        }
        lse2[t] = lse * lse;
    }
}

__global__ __launch_bounds__(256) void moe_losses(const float* __restrict__ probs,
                                                  const float* __restrict__ lse2,
                                                  float* __restrict__ out_loss) {
    __shared__ float red[256];
    float us[NE] = {};
    float zs = 0.0f;
    for (int t = threadIdx.x; t < NTOK; t += 256) {
#pragma unroll
        for (int e = 0; e < NE; ++e) us[e] += probs[t * NE + e];
        zs += lse2[t];
    }
    float totals[NE + 1];
    for (int v = 0; v < NE + 1; ++v) {
        red[threadIdx.x] = (v < NE) ? us[v] : zs;
        __syncthreads();
        for (int s = 128; s > 0; s >>= 1) {
            if (threadIdx.x < (unsigned)s) red[threadIdx.x] += red[threadIdx.x + s];
            __syncthreads();
        }
        totals[v] = red[0];
        __syncthreads();
    }
    if (threadIdx.x == 0) {
        float usage[NE]; float mean = 0.0f;
#pragma unroll
        for (int e = 0; e < NE; ++e) { usage[e] = totals[e] / (float)NTOK; mean += usage[e]; }
        mean *= (1.0f / NE);
        float var = 0.0f;
#pragma unroll
        for (int e = 0; e < NE; ++e) { float d = usage[e] - mean; var += d * d; }
        var *= (1.0f / NE);
        float lb = var / (mean * mean + 1e-8f) * (float)NE;
        out_loss[0] = lb * LB_W;
        out_loss[1] = (totals[NE] / (float)NTOK) * Z_W;
    }
}

// one wave per expert: ballot prefix-scan builds token list in token order
__global__ __launch_bounds__(256) void moe_gather(const float* __restrict__ combine,
                                                  int* __restrict__ glist,
                                                  int* __restrict__ counts) {
    const int e = threadIdx.x >> 5;
    const int lane = threadIdx.x & 31;
    int base = 0;
    for (int t0 = 0; t0 < NTOK; t0 += 32) {
        int t = t0 + lane;
        bool sel = combine[t * NE + e] > 0.0f;
        unsigned mask = (unsigned)__ballot(sel);
        if (sel) {
            int pre = __popc(mask & ((1u << lane) - 1u));
            glist[e * NTOK + base + pre] = t;
        }
        base += __popc(mask);
    }
    if (lane == 0) counts[e] = base;
}

// ---------------- WMMA GEMM kernels: 128x128x32 tiles, 8 waves (64x32 each),
// ---------------- double-buffered LDS, async A fills ----------------

// B tile fill (fp32 [KxN] -> bf16 transposed Bt[col][k]); overlaps with WMMA of prev stage.
__device__ __forceinline__ void fill_b_tile(unsigned short* Bt, const float* __restrict__ Wsrc,
                                            int ldw, int k0, int n0, int tid) {
    const int bcol = tid & 127;          // consecutive lanes -> consecutive cols (coalesced)
    const int bkq0 = tid >> 7;           // 0..1
#pragma unroll
    for (int it = 0; it < 2; ++it) {
        int kq = bkq0 + 2 * it;          // K chunk of 8
        const float* wp = Wsrc + (size_t)(k0 + kq * 8) * ldw + n0 + bcol;
        unsigned int u[4];
#pragma unroll
        for (int i = 0; i < 4; ++i) {
            float w0 = wp[(size_t)(2 * i) * ldw];
            float w1 = wp[(size_t)(2 * i + 1) * ldw];
            u[i] = f2bf2(w0, w1);
        }
        *(v4u*)(Bt + bcol * BS_LD + kq * 8) = (v4u){u[0], u[1], u[2], u[3]};
    }
}

// GEMM1: h = gelu(gather(xbf) @ W1[e] + b1[e]) -> bf16
__global__ __launch_bounds__(256) void moe_ffn1(const unsigned short* __restrict__ xbf,
                                                const float* __restrict__ W1e,
                                                const float* __restrict__ b1e,
                                                const int* __restrict__ glist,
                                                const int* __restrict__ cnt_p,
                                                unsigned short* __restrict__ hbuf) {
    __shared__ __align__(16) unsigned short As[2][128 * AS_LD];
    __shared__ __align__(16) unsigned short Bt[2][128 * BS_LD];
    const int cnt = *cnt_p;
    const int m0 = blockIdx.y * 128;
    if (m0 >= cnt) return;
    const int n0 = blockIdx.x * 128;
    const int tid = threadIdx.x;
    const int lane = tid & 31, wid = tid >> 5;
    const int wm = (wid >> 2) * 64, wn = (wid & 3) * 32;

    // A fill: 128 rows x 4 K-chunks(8 bf16) = 512 b128 copies, 2 per thread.
    const int akq = tid & 3;
    const int ar0 = tid >> 2;
    int toks[2];
#pragma unroll
    for (int it = 0; it < 2; ++it) {
        int gi = min(m0 + ar0 + 64 * it, cnt - 1);   // clamp: garbage rows masked at store
        toks[it] = glist[gi];
    }

    v8f cc[4][2] = {};
    const int nk = ND / 32;

    // prologue: stage 0
#pragma unroll
    for (int it = 0; it < 2; ++it)
        async_copy_b128(xbf + (size_t)toks[it] * ND + akq * 8,
                        &As[0][(ar0 + 64 * it) * AS_LD + akq * 8]);
    fill_b_tile(Bt[0], W1e, NF, 0, n0, tid);
    wait_async();
    __syncthreads();

    for (int ks = 0; ks < nk; ++ks) {
        const int cur = ks & 1;
        if (ks + 1 < nk) {
            const int k1 = (ks + 1) * 32;
#pragma unroll
            for (int it = 0; it < 2; ++it)
                async_copy_b128(xbf + (size_t)toks[it] * ND + k1 + akq * 8,
                                &As[cur ^ 1][(ar0 + 64 * it) * AS_LD + akq * 8]);
            fill_b_tile(Bt[cur ^ 1], W1e, NF, k1, n0, tid);
        }
        v16bf a[4], b[2];
#pragma unroll
        for (int mi = 0; mi < 4; ++mi) a[mi] = load_a_frag(As[cur], wm + mi * 16, lane);
#pragma unroll
        for (int nj = 0; nj < 2; ++nj) b[nj] = load_b_frag(Bt[cur], wn + nj * 16, lane);
#pragma unroll
        for (int mi = 0; mi < 4; ++mi)
#pragma unroll
            for (int nj = 0; nj < 2; ++nj)
                cc[mi][nj] = __builtin_amdgcn_wmma_f32_16x16x32_bf16(
                    false, a[mi], false, b[nj], (short)0, cc[mi][nj], false, false);
        wait_async();
        __syncthreads();
    }

    const int half = lane >> 4, r = lane & 15;
#pragma unroll
    for (int mi = 0; mi < 4; ++mi) {
#pragma unroll
        for (int nj = 0; nj < 2; ++nj) {
            int colg = n0 + wn + nj * 16 + r;
            float bias = b1e[colg];
#pragma unroll
            for (int g = 0; g < 8; ++g) {
                int rowg = m0 + wm + mi * 16 + g + 8 * half;
                if (rowg < cnt) {
                    float v = cc[mi][nj][g] + bias;
                    float gl = 0.5f * v * (1.0f + erff(v * 0.70710678118f));
                    hbuf[(size_t)rowg * NF + colg] = f2bf(gl);
                }
            }
        }
    }
}

// GEMM2: out[t] += combine[t][e] * (h @ W2[e] + b2[e])
__global__ __launch_bounds__(256) void moe_ffn2(const unsigned short* __restrict__ hbuf,
                                                const float* __restrict__ W2e,
                                                const float* __restrict__ b2e,
                                                const int* __restrict__ glist,
                                                const int* __restrict__ cnt_p,
                                                const float* __restrict__ combine,
                                                float* __restrict__ out, int e) {
    __shared__ __align__(16) unsigned short As[2][128 * AS_LD];
    __shared__ __align__(16) unsigned short Bt[2][128 * BS_LD];
    const int cnt = *cnt_p;
    const int m0 = blockIdx.y * 128;
    if (m0 >= cnt) return;
    const int n0 = blockIdx.x * 128;
    const int tid = threadIdx.x;
    const int lane = tid & 31, wid = tid >> 5;
    const int wm = (wid >> 2) * 64, wn = (wid & 3) * 32;

    const int akq = tid & 3;
    const int ar0 = tid >> 2;
    int arow[2];
#pragma unroll
    for (int it = 0; it < 2; ++it) arow[it] = min(m0 + ar0 + 64 * it, cnt - 1);

    v8f cc[4][2] = {};
    const int nk = NF / 32;

#pragma unroll
    for (int it = 0; it < 2; ++it)
        async_copy_b128(hbuf + (size_t)arow[it] * NF + akq * 8,
                        &As[0][(ar0 + 64 * it) * AS_LD + akq * 8]);
    fill_b_tile(Bt[0], W2e, ND, 0, n0, tid);
    wait_async();
    __syncthreads();

    for (int ks = 0; ks < nk; ++ks) {
        const int cur = ks & 1;
        if (ks + 1 < nk) {
            const int k1 = (ks + 1) * 32;
#pragma unroll
            for (int it = 0; it < 2; ++it)
                async_copy_b128(hbuf + (size_t)arow[it] * NF + k1 + akq * 8,
                                &As[cur ^ 1][(ar0 + 64 * it) * AS_LD + akq * 8]);
            fill_b_tile(Bt[cur ^ 1], W2e, ND, k1, n0, tid);
        }
        v16bf a[4], b[2];
#pragma unroll
        for (int mi = 0; mi < 4; ++mi) a[mi] = load_a_frag(As[cur], wm + mi * 16, lane);
#pragma unroll
        for (int nj = 0; nj < 2; ++nj) b[nj] = load_b_frag(Bt[cur], wn + nj * 16, lane);
#pragma unroll
        for (int mi = 0; mi < 4; ++mi)
#pragma unroll
            for (int nj = 0; nj < 2; ++nj)
                cc[mi][nj] = __builtin_amdgcn_wmma_f32_16x16x32_bf16(
                    false, a[mi], false, b[nj], (short)0, cc[mi][nj], false, false);
        wait_async();
        __syncthreads();
    }

    const int half = lane >> 4, r = lane & 15;
#pragma unroll
    for (int mi = 0; mi < 4; ++mi) {
#pragma unroll
        for (int nj = 0; nj < 2; ++nj) {
            int colg = n0 + wn + nj * 16 + r;
            float bias = b2e[colg];
#pragma unroll
            for (int g = 0; g < 8; ++g) {
                int rowg = m0 + wm + mi * 16 + g + 8 * half;
                if (rowg < cnt) {
                    int t = glist[rowg];
                    float w = combine[t * NE + e];
                    out[(size_t)t * ND + colg] += w * (cc[mi][nj][g] + bias);
                }
            }
        }
    }
}

// ---------------- launcher ----------------
extern "C" void kernel_launch(void* const* d_in, const int* in_sizes, int n_in,
                              void* d_out, int out_size, void* d_ws, size_t ws_size,
                              hipStream_t stream) {
    const float* x  = (const float*)d_in[0];
    const float* Wr = (const float*)d_in[1];
    const float* W1 = (const float*)d_in[2];
    const float* b1 = (const float*)d_in[3];
    const float* W2 = (const float*)d_in[4];
    const float* b2 = (const float*)d_in[5];
    float* out = (float*)d_out;

    char* ws = (char*)d_ws;
    const size_t OFF_PROBS   = 0;
    const size_t OFF_COMBINE = OFF_PROBS   + (size_t)NTOK * NE * 4;
    const size_t OFF_LSE2    = OFF_COMBINE + (size_t)NTOK * NE * 4;
    const size_t OFF_COUNTS  = OFF_LSE2    + (size_t)NTOK * 4;
    const size_t OFF_GLIST   = OFF_COUNTS  + 64;
    const size_t OFF_XBF     = OFF_GLIST   + (size_t)NE * NTOK * 4;
    const size_t OFF_H       = OFF_XBF     + (size_t)NTOK * ND * 2;

    float* probs   = (float*)(ws + OFF_PROBS);
    float* combine = (float*)(ws + OFF_COMBINE);
    float* lse2    = (float*)(ws + OFF_LSE2);
    int*   counts  = (int*)(ws + OFF_COUNTS);
    int*   glist   = (int*)(ws + OFF_GLIST);
    unsigned short* xbf  = (unsigned short*)(ws + OFF_XBF);  // 8192*768  bf16
    unsigned short* hbuf = (unsigned short*)(ws + OFF_H);    // 8192*3072 bf16

    const int n_out = NTOK * ND;
    moe_zero<<<(n_out + 255) / 256, 256, 0, stream>>>(out, n_out);
    const int nquads = NTOK * ND / 4;
    moe_cvt_x<<<(nquads + 255) / 256, 256, 0, stream>>>(x, xbf, nquads);
    moe_router<<<NTOK / 8, 256, 0, stream>>>(x, Wr, probs, combine, lse2);
    moe_losses<<<1, 256, 0, stream>>>(probs, lse2, out + (size_t)n_out);
    moe_gather<<<1, 256, 0, stream>>>(combine, glist, counts);

    for (int e = 0; e < NE; ++e) {
        dim3 g1(NF / 128, NTOK / 128);   // (24, 64)
        moe_ffn1<<<g1, 256, 0, stream>>>(xbf, W1 + (size_t)e * ND * NF, b1 + (size_t)e * NF,
                                         glist + (size_t)e * NTOK, counts + e, hbuf);
        dim3 g2(ND / 128, NTOK / 128);   // (6, 64)
        moe_ffn2<<<g2, 256, 0, stream>>>(hbuf, W2 + (size_t)e * NF * ND, b2 + (size_t)e * ND,
                                         glist + (size_t)e * NTOK, counts + e, combine, out, e);
    }
}
